// SelfAttentionLayer_65481071401986
// MI455X (gfx1250) — matrix operand order repaired and checked by reading, not compiled, not run on hardware
//
#include <hip/hip_runtime.h>

// ---------------------------------------------------------------------------
// SelfAttention (B=8, N=4096, D=512), fp32 in/out, bf16 WMMA compute.
// Pipeline:
//   1) transpose_bf16: rotation,entangle -> Wrt,Wet (bf16, [e][d]); x -> Vt (bf16, [b][d][n])
//   2) proj_kernel:    Qb = x@rotation, Kb = x@entangle  (v_wmma_f32_16x16x32_bf16)
//   3) fa_kernel:      flash attention; K/V tiles staged with
//                      GLOBAL_LOAD_ASYNC_TO_LDS_B128 (ASYNCcnt) per CDNA5 ISA.
// ---------------------------------------------------------------------------

typedef __attribute__((ext_vector_type(16))) __bf16          v16bf;
typedef __attribute__((ext_vector_type(8)))  float           v8f;
typedef __attribute__((ext_vector_type(4)))  unsigned int    u32x4;
typedef __attribute__((ext_vector_type(4)))  float           f32x4;
typedef __attribute__((ext_vector_type(16))) unsigned short  u16x16;

struct U32x8 { u32x4 lo, hi; };
static_assert(sizeof(U32x8) == 32 && sizeof(v16bf) == 32, "frag size");

__device__ __forceinline__ unsigned short f2bf(float f) {
  unsigned int u = __float_as_uint(f);
  u += 0x7FFFu + ((u >> 16) & 1u);            // round-to-nearest-even
  return (unsigned short)(u >> 16);
}

// 16 contiguous bf16 (contiguous k order; unused in this file)
__device__ __forceinline__ v16bf ld_frag16(const unsigned short* p) {
  U32x8 t;
  t.lo = *reinterpret_cast<const u32x4*>(p);
  t.hi = *reinterpret_cast<const u32x4*>(p + 8);
  return __builtin_bit_cast(v16bf, t);
}

// two 8-element chunks (A-matrix fragment: K = base..+7 and base+16..+23)
__device__ __forceinline__ v16bf ld_frag_2chunk(const unsigned short* p0,
                                                const unsigned short* p1) {
  U32x8 t;
  t.lo = *reinterpret_cast<const u32x4*>(p0);
  t.hi = *reinterpret_cast<const u32x4*>(p1);
  return __builtin_bit_cast(v16bf, t);
}

// CDNA5 async global->LDS copy of 16 bytes (tracked by ASYNCcnt, no VGPR data)
__device__ __forceinline__ void async_g2l_b128(void* lds_ptr, const void* gptr) {
  unsigned lds_off = (unsigned)(size_t)lds_ptr;            // flat->LDS: low 32 bits
  unsigned long long ga = (unsigned long long)(size_t)gptr;
  asm volatile("global_load_async_to_lds_b128 %0, %1, off"
               :: "v"(lds_off), "v"(ga) : "memory");
}

__device__ __forceinline__ void wait_async0() {
  asm volatile("s_wait_asynccnt 0x0" ::: "memory");
}

// ---------------------------------------------------------------------------
// Transpose + fp32->bf16 convert: out[b][c][r] = bf16(in[b][r][c])
// ---------------------------------------------------------------------------
__global__ __launch_bounds__(256) void transpose_bf16_kernel(
    const float* __restrict__ in, unsigned short* __restrict__ outp,
    int R, int C, size_t inStride, size_t outStride) {
  __shared__ unsigned short tileT[64 * 72];
  const int b  = blockIdx.z;
  const int r0 = blockIdx.y * 64;
  const int c0 = blockIdx.x * 64;
  const float* src = in + (size_t)b * inStride;
  unsigned short* dst = outp + (size_t)b * outStride;
  for (int i = 0; i < 16; ++i) {
    int e  = threadIdx.x + i * 256;
    int rr = e >> 6, cc = e & 63;
    tileT[cc * 72 + rr] = f2bf(src[(size_t)(r0 + rr) * C + c0 + cc]);
  }
  __syncthreads();
  for (int i = 0; i < 16; ++i) {
    int e  = threadIdx.x + i * 256;
    int rr = e >> 6, cc = e & 63;
    dst[(size_t)(c0 + rr) * R + r0 + cc] = tileT[rr * 72 + cc];
  }
}

// ---------------------------------------------------------------------------
// Q/K projection: one wave per 16x16 output tile, shared A (x) fragment
// feeds two WMMAs (rotation + entangle). Weights pre-transposed [e][d].
// ---------------------------------------------------------------------------
__global__ __launch_bounds__(256) void proj_kernel(
    const float* __restrict__ x,
    const unsigned short* __restrict__ Wrt,
    const unsigned short* __restrict__ Wet,
    unsigned short* __restrict__ Qb,
    unsigned short* __restrict__ Kb) {
  constexpr int D = 512, N = 4096;
  const int lane = threadIdx.x & 31;
  const int wave = threadIdx.x >> 5;
  const int l15 = lane & 15, lh = lane >> 4;

  int job = blockIdx.x * 8 + wave;     // 8*256*32 = 65536 jobs
  int b   = job >> 13;
  int rem = job & 8191;
  int q0  = (rem >> 5) * 16;
  int e0  = (rem & 31) * 16;

  v8f accq = {}, acck = {};
  const float*          xrow = x   + ((size_t)b * N + q0 + l15) * D;
  const unsigned short* wr   = Wrt + (size_t)(e0 + l15) * D;
  const unsigned short* we   = Wet + (size_t)(e0 + l15) * D;

  for (int kk = 0; kk < D; kk += 32) {
    const float* p0 = xrow + kk + lh * 8;
    const float* p1 = p0 + 16;
    f32x4 f0 = *(const f32x4*)(p0);
    f32x4 f1 = *(const f32x4*)(p0 + 4);
    f32x4 f2 = *(const f32x4*)(p1);
    f32x4 f3 = *(const f32x4*)(p1 + 4);
    u16x16 ua;
#pragma unroll
    for (int i = 0; i < 4; ++i) {
      ua[i]      = f2bf(f0[i]);
      ua[4 + i]  = f2bf(f1[i]);
      ua[8 + i]  = f2bf(f2[i]);
      ua[12 + i] = f2bf(f3[i]);
    }
    v16bf a  = __builtin_bit_cast(v16bf, ua);
    v16bf bq = ld_frag_2chunk(wr + kk + lh * 8, wr + kk + 16 + lh * 8);
    v16bf bk = ld_frag_2chunk(we + kk + lh * 8, we + kk + 16 + lh * 8);
    accq = __builtin_amdgcn_wmma_f32_16x16x32_bf16(false, a, false, bq, (short)0, accq, false, false);
    acck = __builtin_amdgcn_wmma_f32_16x16x32_bf16(false, a, false, bk, (short)0, acck, false, false);
  }
#pragma unroll
  for (int i = 0; i < 8; ++i) {
    size_t idx = ((size_t)b * N + q0 + i + lh * 8) * D + e0 + l15;
    Qb[idx] = f2bf(accq[i]);
    Kb[idx] = f2bf(acck[i]);
  }
}

// ---------------------------------------------------------------------------
// Flash attention: 512 threads = 16 waves as a 4x4 grid.
//  wave (r,c): S-tile rows [16r,16r+16) x keys [16c,16c+16)
//              O rows [16r,16r+16) x dims [128c,128c+128)
// ---------------------------------------------------------------------------
__global__ __launch_bounds__(512) void fa_kernel(
    const unsigned short* __restrict__ Qb,
    const unsigned short* __restrict__ Kb,
    const unsigned short* __restrict__ Vt,
    float* __restrict__ out) {
  constexpr int D = 512, N = 4096;
  constexpr int QBL = 64, KBL = 64;
  constexpr int QS = D + 8;      // sQ/sK row stride (ushorts)
  constexpr int VS = KBL + 8;    // sV/sP row stride (ushorts)

  __shared__ unsigned short sQ[QBL * QS];
  __shared__ unsigned short sK[KBL * QS];
  __shared__ unsigned short sV[D * VS];     // [dim][key]
  __shared__ unsigned short sP[QBL * VS];   // [q][key]
  __shared__ float sRmax[4][QBL];
  __shared__ float sRsum[4][QBL];

  const int tid  = threadIdx.x;
  const int lane = tid & 31;
  const int wave = tid >> 5;
  const int r = wave >> 2, c = wave & 3;
  const int l15 = lane & 15, lh = lane >> 4;

  const int b  = blockIdx.y;
  const int q0 = blockIdx.x * QBL;
  const float SCL = 1.44269504089f / sqrtf((float)D);  // log2(e)/sqrt(D)

  // Preload Q tile (64 x 512 bf16) via async global->LDS
  {
    const unsigned short* src = Qb + ((size_t)b * N + q0) * D;
    for (int g = tid; g < (QBL * D) / 8; g += 512) {
      int row = g >> 6, col8 = (g & 63) * 8;
      async_g2l_b128(&sQ[row * QS + col8], src + row * D + col8);
    }
  }

  v8f vzero = {};
  float m[8], lsum[8], corr[8];
  v8f o[8];
#pragma unroll
  for (int i = 0; i < 8; ++i) { m[i] = -1e30f; lsum[i] = 0.f; o[i] = vzero; }

  const unsigned short* kbase = Kb + (size_t)b * N * D;
  const unsigned short* vbase = Vt + (size_t)b * D * N;

  for (int kb = 0; kb < N / KBL; ++kb) {
    const int k0 = kb * KBL;
    // --- stage K/V tiles into LDS (async, tracked by ASYNCcnt) ---
    {
      const unsigned short* src = kbase + (size_t)k0 * D;
      for (int g = tid; g < (KBL * D) / 8; g += 512) {
        int row = g >> 6, col8 = (g & 63) * 8;
        async_g2l_b128(&sK[row * QS + col8], src + row * D + col8);
        if (kb + 1 < N / KBL)
          __builtin_prefetch(src + (size_t)KBL * D + row * D + col8, 0, 1);
      }
      for (int g = tid; g < (D * KBL) / 8; g += 512) {
        int dim = g >> 3, col8 = (g & 7) * 8;
        async_g2l_b128(&sV[dim * VS + col8], vbase + (size_t)dim * N + k0 + col8);
        if (kb + 1 < N / KBL)
          __builtin_prefetch(vbase + (size_t)dim * N + k0 + KBL + col8, 0, 1);
      }
    }
    wait_async0();       // drain this wave's async LDS writes (covers Q preload too)
    __syncthreads();     // all waves' writes now visible

    // --- S = Q K^T tile (16 WMMAs over K=512) ---
    v8f s = vzero;
    {
      const unsigned short* qrow = &sQ[(16 * r + l15) * QS];
      const unsigned short* krow = &sK[(16 * c + l15) * QS];
      for (int kk = 0; kk < D; kk += 32) {
        v16bf a  = ld_frag_2chunk(qrow + kk + lh * 8, qrow + kk + 16 + lh * 8);
        v16bf bf = ld_frag_2chunk(krow + kk + lh * 8, krow + kk + 16 + lh * 8);
        s = __builtin_amdgcn_wmma_f32_16x16x32_bf16(false, a, false, bf, (short)0, s, false, false);
      }
    }

    // --- per-row max: intra-wave reduce + cross-wave via LDS ---
#pragma unroll
    for (int i = 0; i < 8; ++i) {
      float v = s[i];
      v = fmaxf(v, __shfl_xor(v, 8, 32));
      v = fmaxf(v, __shfl_xor(v, 4, 32));
      v = fmaxf(v, __shfl_xor(v, 2, 32));
      v = fmaxf(v, __shfl_xor(v, 1, 32));
      if (l15 == 0) sRmax[c][16 * r + i + lh * 8] = v;
    }
    __syncthreads();

    // --- online softmax: P = exp2((s-m)*SCL), write bf16 P to LDS ---
#pragma unroll
    for (int i = 0; i < 8; ++i) {
      int rr = 16 * r + i + lh * 8;
      float bm = fmaxf(fmaxf(sRmax[0][rr], sRmax[1][rr]),
                       fmaxf(sRmax[2][rr], sRmax[3][rr]));
      float mn = fmaxf(m[i], bm);
      corr[i] = exp2f((m[i] - mn) * SCL);
      m[i] = mn;
      float p = exp2f((s[i] - mn) * SCL);
      sP[rr * VS + 16 * c + l15] = f2bf(p);
      float ps = p;
      ps += __shfl_xor(ps, 8, 32);
      ps += __shfl_xor(ps, 4, 32);
      ps += __shfl_xor(ps, 2, 32);
      ps += __shfl_xor(ps, 1, 32);
      if (l15 == 0) sRsum[c][rr] = ps;
    }
    // rescale O accumulators by correction factor
#pragma unroll
    for (int t = 0; t < 8; ++t)
#pragma unroll
      for (int i = 0; i < 8; ++i) o[t][i] *= corr[i];
    __syncthreads();

#pragma unroll
    for (int i = 0; i < 8; ++i) {
      int rr = 16 * r + i + lh * 8;
      lsum[i] = lsum[i] * corr[i] +
                (sRsum[0][rr] + sRsum[1][rr] + sRsum[2][rr] + sRsum[3][rr]);
    }

    // --- O += P * V  (8 dim-tiles x 2 K-steps = 16 WMMAs) ---
    {
      const unsigned short* prow = &sP[(16 * r + l15) * VS];
#pragma unroll
      for (int t = 0; t < 8; ++t) {
        const unsigned short* vrow = &sV[(128 * c + 16 * t + l15) * VS];
#pragma unroll
        for (int k2 = 0; k2 < 2; ++k2) {
          v16bf a  = ld_frag_2chunk(prow + k2 * 32 + lh * 8,
                                    prow + k2 * 32 + 16 + lh * 8);
          v16bf bf = ld_frag_2chunk(vrow + k2 * 32 + lh * 8, vrow + k2 * 32 + 16 + lh * 8);
          o[t] = __builtin_amdgcn_wmma_f32_16x16x32_bf16(false, a, false, bf, (short)0, o[t], false, false);
        }
      }
    }
    __syncthreads();   // before overwriting sK/sV next iteration
  }

  // --- normalize and store fp32 output ---
#pragma unroll
  for (int i = 0; i < 8; ++i) {
    float inv = 1.0f / lsum[i];
    int row = q0 + 16 * r + i + lh * 8;
    float* op = out + ((size_t)b * N + row) * D + 128 * c + l15;
#pragma unroll
    for (int t = 0; t < 8; ++t) op[16 * t] = o[t][i] * inv;
  }
}

// ---------------------------------------------------------------------------
extern "C" void kernel_launch(void* const* d_in, const int* in_sizes, int n_in,
                              void* d_out, int out_size, void* d_ws, size_t ws_size,
                              hipStream_t stream) {
  (void)in_sizes; (void)n_in; (void)out_size; (void)ws_size;
  constexpr int Bn = 8, N = 4096, D = 512;
  const float* x        = (const float*)d_in[0];
  const float* rotation = (const float*)d_in[1];
  const float* entangle = (const float*)d_in[2];
  float* out = (float*)d_out;

  unsigned short* Wrt = (unsigned short*)d_ws;                 //  512x512
  unsigned short* Wet = Wrt + (size_t)D * D;                   //  512x512
  unsigned short* Qb  = Wet + (size_t)D * D;                   //  8x4096x512
  unsigned short* Kb  = Qb + (size_t)Bn * N * D;               //  8x4096x512
  unsigned short* Vt  = Kb + (size_t)Bn * N * D;               //  8x512x4096

  // Wrt[e][d] = rotation[d][e]; Wet likewise
  transpose_bf16_kernel<<<dim3(D / 64, D / 64, 1), 256, 0, stream>>>(
      rotation, Wrt, D, D, 0, 0);
  transpose_bf16_kernel<<<dim3(D / 64, D / 64, 1), 256, 0, stream>>>(
      entangle, Wet, D, D, 0, 0);
  // Vt[b][d][n] = bf16(x[b][n][d])
  transpose_bf16_kernel<<<dim3(D / 64, N / 64, Bn), 256, 0, stream>>>(
      x, Vt, N, D, (size_t)N * D, (size_t)D * N);
  // Q/K projections
  proj_kernel<<<(Bn * (N / 16) * (D / 16)) / 8, 256, 0, stream>>>(
      x, Wrt, Wet, Qb, Kb);
  // fused flash attention
  fa_kernel<<<dim3(N / 64, Bn), 512, 0, stream>>>(Qb, Kb, Vt, out);
}
